// DbrxAttention_10574209483530
// MI455X (gfx1250) — compile-verified, matrix operand order checked by reading, not run on hardware
//
#include <hip/hip_runtime.h>
#include <hip/hip_bf16.h>

// ---------------------------------------------------------------------------
// DBRX attention block on gfx1250 (MI455X): bf16 WMMA pipeline, f32 accum.
// Stages: pack -> QKV GEMM(+clip) -> RoPE(+scale folded into Q) -> flash attn
//         -> output GEMM. Matrix math via v_wmma_f32_16x16x32_bf16; tile
//         staging via global_load_async_to_lds_b128 (ASYNCcnt).
// ---------------------------------------------------------------------------

typedef __attribute__((ext_vector_type(16))) __bf16 v16bf;
typedef __attribute__((ext_vector_type(8)))  __bf16 v8bf;
typedef __attribute__((ext_vector_type(8)))  float  v8f;

#define D_MODEL 6144
#define N_HEADS 48
#define N_KV    8
#define HEAD    128
#define SEQ     1024
#define BATCH   4
#define TOKENS  (BATCH * SEQ)                       // 4096
#define QKV_N   ((N_HEADS + 2 * N_KV) * HEAD)       // 8192
#define GQA_G   (N_HEADS / N_KV)                    // 6
#define CLIPV   8.0f
#define SCALE   0.08838834764831845f                // 128^-0.5

__device__ __forceinline__ v8f wmma_bf16(v16bf a, v16bf b, v8f c) {
    // 8-arg form: (neg_a, A, neg_b, B, c_mod, C, reuse_a, reuse_b)
    return __builtin_amdgcn_wmma_f32_16x16x32_bf16(false, a, false, b,
                                                   (short)0, c, false, false);
}

// CDNA5 async copy: global memory -> LDS, 16 bytes per lane, ASYNCcnt-tracked.
// VDST operand = LDS byte address in a VGPR (flat low-32 == LDS offset).
__device__ __forceinline__ void async_ld_b128(void* lds_ptr, const void* gptr) {
    unsigned lds = (unsigned)(size_t)lds_ptr;
    asm volatile("global_load_async_to_lds_b128 %0, %1, off"
                 :: "v"(lds), "v"(gptr) : "memory");
}
__device__ __forceinline__ void wait_async0() {
    asm volatile("s_wait_asynccnt 0" ::: "memory");
}

// Load one 16-bf16 WMMA operand fragment from an LDS row.
// CDNA5 16-bit A/B layout: lanes 0-15 hold K = koff..koff+7 and koff+16..+23
// with koff = 0; lanes 16-31 with koff = 8.  Caller passes koff = (lane>>4)*8.
__device__ __forceinline__ v16bf load_frag(const __bf16* rowp, int koff) {
    v8bf lo = *(const v8bf*)(rowp + koff);
    v8bf hi = *(const v8bf*)(rowp + koff + 16);
    return __builtin_shufflevector(lo, hi, 0, 1, 2, 3, 4, 5, 6, 7,
                                           8, 9, 10, 11, 12, 13, 14, 15);
}

__device__ __forceinline__ float half_reduce_max(float v) {
    v = fmaxf(v, __shfl_xor(v, 1, 32));
    v = fmaxf(v, __shfl_xor(v, 2, 32));
    v = fmaxf(v, __shfl_xor(v, 4, 32));
    v = fmaxf(v, __shfl_xor(v, 8, 32));
    return v;  // reduced within each 16-lane half (rows 0-7 / 8-15)
}
__device__ __forceinline__ float half_reduce_sum(float v) {
    v += __shfl_xor(v, 1, 32);
    v += __shfl_xor(v, 2, 32);
    v += __shfl_xor(v, 4, 32);
    v += __shfl_xor(v, 8, 32);
    return v;
}

// ---------------------------------------------------------------------------
// Pack: f32 -> bf16, flat grid-stride
// ---------------------------------------------------------------------------
__global__ void pack_bf16_kernel(const float* __restrict__ X,
                                 __bf16* __restrict__ Y, long n) {
    long i = (long)blockIdx.x * blockDim.x + threadIdx.x;
    long stride = (long)gridDim.x * blockDim.x;
    for (; i < n; i += stride) Y[i] = (__bf16)X[i];
}

// ---------------------------------------------------------------------------
// Pack + transpose: W[K][N] f32 -> Wt[N][K] bf16 (32x32 LDS tiles)
// ---------------------------------------------------------------------------
__global__ __launch_bounds__(256)
void transpose_pack_kernel(const float* __restrict__ W, __bf16* __restrict__ Wt,
                           int K, int N) {
    __shared__ __bf16 tile[32][33];
    const int n0 = blockIdx.x * 32, k0 = blockIdx.y * 32;
    const int tx = threadIdx.x & 31, ty = threadIdx.x >> 5;  // ty: 0..7
    for (int it = 0; it < 4; ++it) {
        int k = ty + it * 8;
        tile[k][tx] = (__bf16)W[(size_t)(k0 + k) * N + n0 + tx];
    }
    __syncthreads();
    for (int it = 0; it < 4; ++it) {
        int n = ty + it * 8;
        Wt[(size_t)(n0 + n) * K + k0 + tx] = tile[tx][n];
    }
}

// ---------------------------------------------------------------------------
// GEMM: C[M][N] = A[M][K] @ Bt[N][K]^T   (A,Bt bf16; C f32)
// Block tile 128x128, K-step 64, 8 waves, wave tile 64x32 (4x2 WMMA tiles).
// Double-buffered LDS; tiles staged with async global->LDS (ASYNCcnt).
// One barrier per K stage; 16 WMMA per stage per wave.
// ---------------------------------------------------------------------------
#define GEMM_LDSROW 72   // 64 + 8 pad -> conflict-free ds_load_b128
template <bool CLIP_OUT>
__global__ __launch_bounds__(256)
void gemm_bf16_kernel(const __bf16* __restrict__ A, const __bf16* __restrict__ Bt,
                      float* __restrict__ C, int M, int N, int K) {
    extern __shared__ __bf16 gsm[];
    __bf16(*As)[128][GEMM_LDSROW] = (__bf16(*)[128][GEMM_LDSROW])gsm;
    __bf16(*Bs)[128][GEMM_LDSROW] =
        (__bf16(*)[128][GEMM_LDSROW])(gsm + 2 * 128 * GEMM_LDSROW);

    const int tid = threadIdx.x;
    const int m0 = blockIdx.y * 128;
    const int n0 = blockIdx.x * 128;
    const int lane = tid & 31, wave = tid >> 5;
    const int wm = (wave & 1) * 64;        // wave M offset in block
    const int wn = (wave >> 1) * 32;       // wave N offset in block
    const int lrow = lane & 15;
    const int khalf = lane >> 4;
    const int koff = khalf * 8;

    const v8f vzero = {0.f, 0.f, 0.f, 0.f, 0.f, 0.f, 0.f, 0.f};
    v8f acc[4][2];
#pragma unroll
    for (int i = 0; i < 4; ++i)
#pragma unroll
        for (int j = 0; j < 2; ++j) acc[i][j] = vzero;

    // staging map: 128 rows x 8 chunks(8 bf16) = 1024 chunks / 256 thr = 4 ea
    const __bf16* gA = A + (size_t)m0 * K;
    const __bf16* gB = Bt + (size_t)n0 * K;

    // prologue: stage tile k0 = 0 into buffer 0 (async)
#pragma unroll
    for (int it = 0; it < 4; ++it) {
        int chunk = tid + it * 256;
        int r = chunk >> 3, c = (chunk & 7) * 8;
        async_ld_b128(&As[0][r][c], gA + (size_t)r * K + c);
        async_ld_b128(&Bs[0][r][c], gB + (size_t)r * K + c);
    }

    int cur = 0;
    for (int k0 = 0; k0 < K; k0 += 64) {
        wait_async0();       // this wave's staging (into buf cur) complete
        __syncthreads();     // all waves' staging visible; prev readers done
        if (k0 + 64 < K) {   // stage next tile into the other buffer
            int nxt = cur ^ 1;
#pragma unroll
            for (int it = 0; it < 4; ++it) {
                int chunk = tid + it * 256;
                int r = chunk >> 3, c = (chunk & 7) * 8;
                async_ld_b128(&As[nxt][r][c], gA + (size_t)r * K + k0 + 64 + c);
                async_ld_b128(&Bs[nxt][r][c], gB + (size_t)r * K + k0 + 64 + c);
            }
            if (k0 + 128 < K) {
                __builtin_prefetch(gA + (size_t)(tid >> 1) * K + k0 + 128, 0, 1);
                __builtin_prefetch(gB + (size_t)(tid >> 1) * K + k0 + 128, 0, 1);
            }
        }
        // consume buffer cur: 2 k-steps of 32, 8 WMMA each
#pragma unroll
        for (int ks2 = 0; ks2 < 2; ++ks2) {
            const int ko = ks2 * 32 + koff;
            v16bf af[4], bf2[2];
#pragma unroll
            for (int i = 0; i < 4; ++i)
                af[i] = load_frag(&As[cur][wm + i * 16 + lrow][0], ko);
#pragma unroll
            for (int j = 0; j < 2; ++j)
                bf2[j] = load_frag(&Bs[cur][wn + j * 16 + lrow][0], ko);
#pragma unroll
            for (int i = 0; i < 4; ++i)
#pragma unroll
                for (int j = 0; j < 2; ++j)
                    acc[i][j] = wmma_bf16(af[i], bf2[j], acc[i][j]);
        }
        cur ^= 1;
    }

    // Epilogue.  C layout: VGPR r -> row (r + khalf*8), col = lane&15.
#pragma unroll
    for (int i = 0; i < 4; ++i)
#pragma unroll
        for (int j = 0; j < 2; ++j)
#pragma unroll
            for (int r = 0; r < 8; ++r) {
                int row = m0 + wm + i * 16 + r + khalf * 8;
                int col = n0 + wn + j * 16 + lrow;
                float v = acc[i][j][r];
                if (CLIP_OUT) v = fminf(CLIPV, fmaxf(-CLIPV, v));
                C[(size_t)row * N + col] = v;
            }
}

// ---------------------------------------------------------------------------
// RoPE for Q (scale folded in): qkv f32 -> Qg bf16 [B][H][SEQ][HEAD]
// ---------------------------------------------------------------------------
__global__ void rope_q_kernel(const float* __restrict__ qkv,
                              const float* __restrict__ cs,
                              const float* __restrict__ sn,
                              __bf16* __restrict__ Qg) {
    int idx = blockIdx.x * blockDim.x + threadIdx.x;
    int d = idx & 63;
    int h = (idx >> 6) % N_HEADS;
    int t = idx / (64 * N_HEADS);
    if (t >= TOKENS) return;
    float q1 = qkv[(size_t)t * QKV_N + h * HEAD + d];
    float q2 = qkv[(size_t)t * QKV_N + h * HEAD + 64 + d];
    float c = cs[t * 64 + d], s = sn[t * 64 + d];
    float o1 = (q1 * c - q2 * s) * SCALE;
    float o2 = (q2 * c + q1 * s) * SCALE;
    int b = t / SEQ, sq = t % SEQ;
    size_t base = (((size_t)(b * N_HEADS + h)) * SEQ + sq) * HEAD;
    Qg[base + d] = (__bf16)o1;
    Qg[base + 64 + d] = (__bf16)o2;
}

// RoPE for K: qkv f32 -> Kg bf16 [B][KV][SEQ][HEAD]
__global__ void rope_k_kernel(const float* __restrict__ qkv,
                              const float* __restrict__ cs,
                              const float* __restrict__ sn,
                              __bf16* __restrict__ Kg) {
    int idx = blockIdx.x * blockDim.x + threadIdx.x;
    int d = idx & 63;
    int kh = (idx >> 6) % N_KV;
    int t = idx / (64 * N_KV);
    if (t >= TOKENS) return;
    size_t src = (size_t)t * QKV_N + N_HEADS * HEAD + kh * HEAD;
    float k1 = qkv[src + d];
    float k2 = qkv[src + 64 + d];
    float c = cs[t * 64 + d], s = sn[t * 64 + d];
    int b = t / SEQ, sq = t % SEQ;
    size_t base = (((size_t)(b * N_KV + kh)) * SEQ + sq) * HEAD;
    Kg[base + d] = (__bf16)(k1 * c - k2 * s);
    Kg[base + 64 + d] = (__bf16)(k2 * c + k1 * s);
}

// V transpose pack: qkv f32 -> Vtg bf16 [B][KV][HEAD][SEQ]  (B-operand ready)
__global__ void pack_v_kernel(const float* __restrict__ qkv,
                              __bf16* __restrict__ Vtg) {
    int idx = blockIdx.x * blockDim.x + threadIdx.x;
    int d = idx & 127;
    int kh = (idx >> 7) & 7;
    int t = idx / (HEAD * N_KV);
    if (t >= TOKENS) return;
    float v = qkv[(size_t)t * QKV_N + (N_HEADS + N_KV) * HEAD + kh * HEAD + d];
    int b = t / SEQ, sq = t % SEQ;
    Vtg[(((size_t)(b * N_KV + kh)) * HEAD + d) * SEQ + sq] = (__bf16)v;
}

// ---------------------------------------------------------------------------
// Flash attention: one WG per (batch, head, 128-query block). 8 waves; each
// wave owns 16 query rows. Online softmax; QK^T and P*V via bf16 WMMA.
// Q/K/V tiles staged with async global->LDS. Dynamic LDS: 4 x [128][136] bf16.
// ---------------------------------------------------------------------------
__global__ __launch_bounds__(256)
void flash_attn_kernel(const __bf16* __restrict__ Qg, const __bf16* __restrict__ Kg,
                       const __bf16* __restrict__ Vtg, __bf16* __restrict__ Og) {
    extern __shared__ __bf16 smem[];
    __bf16(*Qs)[136] = (__bf16(*)[136])(smem);
    __bf16(*Ks)[136] = (__bf16(*)[136])(smem + 128 * 136);
    __bf16(*Vs)[136] = (__bf16(*)[136])(smem + 2 * 128 * 136);
    __bf16(*Ps)[136] = (__bf16(*)[136])(smem + 3 * 128 * 136);

    const int tid = threadIdx.x;
    const int qb = blockIdx.x;       // query block (0..7)
    const int h = blockIdx.y;        // head
    const int b = blockIdx.z;        // batch
    const int kh = h / GQA_G;        // kv head
    const int lane = tid & 31, wave = tid >> 5;
    const int lrow = lane & 15;
    const int khalf = lane >> 4;
    const int koff = khalf * 8;
    const int qr0 = wave * 16;       // this wave's query rows within the block

    // stage Q block [128][128] (already RoPE'd and pre-scaled), async
    const __bf16* qbase = Qg + (((size_t)(b * N_HEADS + h)) * SEQ + qb * 128) * HEAD;
#pragma unroll
    for (int it = 0; it < 8; ++it) {
        int chunk = tid + it * 256;
        int r = chunk >> 4, c = (chunk & 15) * 8;
        async_ld_b128(&Qs[r][c], qbase + (size_t)r * HEAD + c);
    }

    const v8f vzero = {0.f, 0.f, 0.f, 0.f, 0.f, 0.f, 0.f, 0.f};
    v8f Ot[8];
    float m_run[8], l_run[8];
#pragma unroll
    for (int n = 0; n < 8; ++n) Ot[n] = vzero;
#pragma unroll
    for (int r = 0; r < 8; ++r) { m_run[r] = -1e30f; l_run[r] = 0.f; }

    const __bf16* kbase = Kg + ((size_t)(b * N_KV + kh)) * SEQ * HEAD;
    const __bf16* vbase = Vtg + ((size_t)(b * N_KV + kh)) * HEAD * SEQ;

    for (int kb = 0; kb <= qb; ++kb) {
        __syncthreads();  // prev iteration's readers of Ks/Vs are done
#pragma unroll
        for (int it = 0; it < 8; ++it) {
            int chunk = tid + it * 256;
            int r = chunk >> 4, c = (chunk & 15) * 8;
            async_ld_b128(&Ks[r][c], kbase + ((size_t)(kb * 128 + r)) * HEAD + c);
            async_ld_b128(&Vs[r][c], vbase + (size_t)r * SEQ + kb * 128 + c);
        }
        wait_async0();    // covers Q staging on first iteration too
        __syncthreads();

        // S = Q K^T  (K-dim = HEAD = 4 WMMA k-steps of 32)
        v16bf aq[4];
#pragma unroll
        for (int ks = 0; ks < 4; ++ks)
            aq[ks] = load_frag(&Qs[qr0 + lrow][0], ks * 32 + koff);
        v8f s[8];
#pragma unroll
        for (int j = 0; j < 8; ++j) {
            s[j] = vzero;
#pragma unroll
            for (int ks = 0; ks < 4; ++ks) {
                v16bf bk = load_frag(&Ks[j * 16 + lrow][0], ks * 32 + koff);
                s[j] = wmma_bf16(aq[ks], bk, s[j]);
            }
        }

        if (kb == qb) {  // causal mask within the diagonal block
#pragma unroll
            for (int j = 0; j < 8; ++j)
#pragma unroll
                for (int r = 0; r < 8; ++r) {
                    int col = j * 16 + lrow;
                    int row = qr0 + r + khalf * 8;
                    if (col > row) s[j][r] = -1e30f;
                }
        }

        // online softmax update
        float mnew[8], alpha[8];
#pragma unroll
        for (int r = 0; r < 8; ++r) {
            float mx = s[0][r];
#pragma unroll
            for (int j = 1; j < 8; ++j) mx = fmaxf(mx, s[j][r]);
            mx = half_reduce_max(mx);
            mnew[r] = fmaxf(m_run[r], mx);
            alpha[r] = __expf(m_run[r] - mnew[r]);
            m_run[r] = mnew[r];
        }
#pragma unroll
        for (int j = 0; j < 8; ++j)
#pragma unroll
            for (int r = 0; r < 8; ++r) s[j][r] = __expf(s[j][r] - mnew[r]);
#pragma unroll
        for (int r = 0; r < 8; ++r) {
            float sm = 0.f;
#pragma unroll
            for (int j = 0; j < 8; ++j) sm += s[j][r];
            sm = half_reduce_sum(sm);
            l_run[r] = l_run[r] * alpha[r] + sm;
        }
#pragma unroll
        for (int n = 0; n < 8; ++n)
#pragma unroll
            for (int r = 0; r < 8; ++r) Ot[n][r] *= alpha[r];

        // C-layout -> A-layout via LDS round trip (wave-private rows; in-wave
        // DScnt ordering makes the reload safe without a block barrier)
#pragma unroll
        for (int j = 0; j < 8; ++j)
#pragma unroll
            for (int r = 0; r < 8; ++r)
                Ps[qr0 + r + khalf * 8][j * 16 + lrow] = (__bf16)s[j][r];

        // O += P V  (K-dim = 128 keys = 4 WMMA k-steps)
        v16bf ap[4];
#pragma unroll
        for (int ks = 0; ks < 4; ++ks)
            ap[ks] = load_frag(&Ps[qr0 + lrow][0], ks * 32 + koff);
#pragma unroll
        for (int n = 0; n < 8; ++n)
#pragma unroll
            for (int ks = 0; ks < 4; ++ks) {
                v16bf bv = load_frag(&Vs[n * 16 + lrow][0], ks * 32 + koff);
                Ot[n] = wmma_bf16(ap[ks], bv, Ot[n]);
            }
    }

    // normalize and emit attn output as bf16 [t][h*128+d]
    const int t0 = b * SEQ + qb * 128 + qr0;
#pragma unroll
    for (int n = 0; n < 8; ++n)
#pragma unroll
        for (int r = 0; r < 8; ++r) {
            int row = t0 + r + khalf * 8;
            int col = h * HEAD + n * 16 + lrow;
            Og[(size_t)row * (N_HEADS * HEAD) + col] =
                (__bf16)(Ot[n][r] / l_run[r]);
        }
}

// ---------------------------------------------------------------------------
// Host side
// ---------------------------------------------------------------------------
extern "C" void kernel_launch(void* const* d_in, const int* in_sizes, int n_in,
                              void* d_out, int out_size, void* d_ws, size_t ws_size,
                              hipStream_t stream) {
    const float* hidden = (const float*)d_in[0];
    const float* w_qkv = (const float*)d_in[1];
    const float* w_o = (const float*)d_in[2];
    const float* cosb = (const float*)d_in[3];
    const float* sinb = (const float*)d_in[4];
    float* out = (float*)d_out;

    char* ws = (char*)d_ws;
    size_t off = 0;
    auto alloc = [&](size_t bytes) -> char* {
        char* p = ws + off;
        off += (bytes + 255) & ~(size_t)255;
        return p;
    };
    __bf16* hiddenBf = (__bf16*)alloc((size_t)TOKENS * D_MODEL * 2);
    __bf16* wqkvT = (__bf16*)alloc((size_t)QKV_N * D_MODEL * 2);
    __bf16* woT = (__bf16*)alloc((size_t)D_MODEL * D_MODEL * 2);
    float* qkv = (float*)alloc((size_t)TOKENS * QKV_N * 4);
    __bf16* Qg = (__bf16*)alloc((size_t)TOKENS * N_HEADS * HEAD * 2);
    __bf16* Kg = (__bf16*)alloc((size_t)TOKENS * N_KV * HEAD * 2);
    __bf16* Vtg = (__bf16*)alloc((size_t)TOKENS * N_KV * HEAD * 2);
    __bf16* attnBf = (__bf16*)alloc((size_t)TOKENS * D_MODEL * 2);

    // 1. pack operands to bf16 (weights transposed to [N][K])
    pack_bf16_kernel<<<4096, 256, 0, stream>>>(hidden, hiddenBf,
                                               (long)TOKENS * D_MODEL);
    transpose_pack_kernel<<<dim3(QKV_N / 32, D_MODEL / 32), 256, 0, stream>>>(
        w_qkv, wqkvT, D_MODEL, QKV_N);
    transpose_pack_kernel<<<dim3(D_MODEL / 32, D_MODEL / 32), 256, 0, stream>>>(
        w_o, woT, D_MODEL, D_MODEL);

    // 2. QKV projection with fused clip (double-buffered async LDS GEMM)
    size_t gemm_shmem = (size_t)4 * 128 * GEMM_LDSROW * sizeof(__bf16);  // 73728
    (void)hipFuncSetAttribute((const void*)gemm_bf16_kernel<true>,
                              hipFuncAttributeMaxDynamicSharedMemorySize,
                              (int)gemm_shmem);
    (void)hipFuncSetAttribute((const void*)gemm_bf16_kernel<false>,
                              hipFuncAttributeMaxDynamicSharedMemorySize,
                              (int)gemm_shmem);
    gemm_bf16_kernel<true>
        <<<dim3(QKV_N / 128, TOKENS / 128), 256, gemm_shmem, stream>>>(
            hiddenBf, wqkvT, qkv, TOKENS, QKV_N, D_MODEL);

    // 3. RoPE (+softmax scale into Q) and V transpose pack
    {
        int tq = TOKENS * N_HEADS * 64;
        rope_q_kernel<<<(tq + 255) / 256, 256, 0, stream>>>(qkv, cosb, sinb, Qg);
        int tk = TOKENS * N_KV * 64;
        rope_k_kernel<<<(tk + 255) / 256, 256, 0, stream>>>(qkv, cosb, sinb, Kg);
        int tv = TOKENS * N_KV * HEAD;
        pack_v_kernel<<<(tv + 255) / 256, 256, 0, stream>>>(qkv, Vtg);
    }

    // 4. causal flash attention (dynamic LDS: 4 x 128 x 136 bf16 = 136 KB)
    size_t shmem = (size_t)4 * 128 * 136 * sizeof(__bf16);
    (void)hipFuncSetAttribute((const void*)flash_attn_kernel,
                              hipFuncAttributeMaxDynamicSharedMemorySize,
                              (int)shmem);
    flash_attn_kernel<<<dim3(SEQ / 128, N_HEADS, BATCH), 256, shmem, stream>>>(
        Qg, Kg, Vtg, attnBf);

    // 5. output projection -> f32 result
    gemm_bf16_kernel<false>
        <<<dim3(D_MODEL / 128, TOKENS / 128), 256, gemm_shmem, stream>>>(
            attnBf, woT, out, TOKENS, D_MODEL, D_MODEL);
}
